// EvaluatorNetwork_76063870812685
// MI455X (gfx1250) — matrix operand order repaired and checked by reading, not compiled, not run on hardware
//
#include <hip/hip_runtime.h>

// ---------------------------------------------------------------------------
// Types for CDNA5 WMMA (wave32)
// ---------------------------------------------------------------------------
typedef __attribute__((ext_vector_type(16))) __bf16        v16bf;
typedef __attribute__((ext_vector_type(8)))  float         v8f;
typedef __attribute__((ext_vector_type(8)))  unsigned int  v8u;

#define LRELU(v) ((v) > 0.0f ? (v) : 0.2f * (v))

// ---------------------------------------------------------------------------
// Problem constants
// ---------------------------------------------------------------------------
static constexpr int B_  = 8;
static constexpr int H_  = 128;
static constexpr int W_  = 128;
static constexpr int EMB = 6;
static constexpr int C0  = W_ + EMB;   // 134 input channels to conv0

// ---------------------------------------------------------------------------
// Zero kernel (grid-stride, 16B granules) — clears halo'd activation buffers
// ---------------------------------------------------------------------------
__global__ __launch_bounds__(256)
void zero_kernel(uint4* __restrict__ p, size_t n16) {
    const size_t stride = (size_t)gridDim.x * 256;
    for (size_t i = (size_t)blockIdx.x * 256 + threadIdx.x; i < n16; i += stride)
        p[i] = make_uint4(0u, 0u, 0u, 0u);
}

// ---------------------------------------------------------------------------
// Weight pack: OIHW f32 -> bf16 in exact WMMA B-fragment layout.
//   k = cin*16 + kh*4 + kw ;  k-chunks of 32 = cin pairs.
//   Bp[((ntile*(Cin/2) + cin/2)*32 + lane)*16 + slot],
//   lane = (n&15) | ((cin&1)<<4), slot = khkw (K ascending within lane).
// One thread per (n, cin): 16 contiguous f32 in, one aligned 32B store out.
// ---------------------------------------------------------------------------
__global__ __launch_bounds__(256)
void pack_weights_kernel(const float* __restrict__ Wt, __bf16* __restrict__ Bp,
                         int Cin, int total) {
    const int idx = blockIdx.x * 256 + threadIdx.x;
    if (idx >= total) return;
    const int n = idx / Cin, cin = idx % Cin;
    const float4* wp = reinterpret_cast<const float4*>(Wt + (size_t)idx * 16);
    const float4 a = wp[0], b = wp[1], c = wp[2], d = wp[3];
    v16bf v;
    v[0]  = (__bf16)a.x; v[1]  = (__bf16)a.y; v[2]  = (__bf16)a.z; v[3]  = (__bf16)a.w;
    v[4]  = (__bf16)b.x; v[5]  = (__bf16)b.y; v[6]  = (__bf16)b.z; v[7]  = (__bf16)b.w;
    v[8]  = (__bf16)c.x; v[9]  = (__bf16)c.y; v[10] = (__bf16)c.z; v[11] = (__bf16)c.w;
    v[12] = (__bf16)d.x; v[13] = (__bf16)d.y; v[14] = (__bf16)d.z; v[15] = (__bf16)d.w;
    const size_t frag = ((size_t)(n >> 4) * (size_t)(Cin >> 1) + (size_t)(cin >> 1)) * 32
                      + (size_t)((n & 15) | ((cin & 1) << 4));
    *reinterpret_cast<v16bf*>(Bp + frag * 16) = v;
}

// ---------------------------------------------------------------------------
// Kernel: spectral map.  sep[b,i,h,w] = (1/W)(C cos(2pi wi/W) + S sin(2pi wi/W)),
// (C,S) = length-128 DFT of image row; exact integer phase table (i*w)&127.
// Writes bf16 into halo'd X0pad (B, 134, 130, 130), channels 0..127.
// ---------------------------------------------------------------------------
__global__ __launch_bounds__(128)
void spectral_map_kernel(const float* __restrict__ image, __bf16* __restrict__ X0pad) {
    constexpr int PH = H_ + 2, PW = W_ + 2;
    const int tid = threadIdx.x;
    const int bh  = blockIdx.x;
    const int b   = bh / H_;
    const int h   = bh % H_;

    __shared__ float row[W_];
    __shared__ float ct[W_], st[W_];
    __shared__ float cbuf[W_], sbuf[W_];

    row[tid] = image[(size_t)(b * H_ + h) * W_ + tid];
    ct[tid]  = __cosf(6.28318530717958647692f * (float)tid / (float)W_);
    st[tid]  = __sinf(6.28318530717958647692f * (float)tid / (float)W_);
    __syncthreads();

    float c = 0.0f, s = 0.0f;
    #pragma unroll 8
    for (int w = 0; w < W_; ++w) {
        const int p = (tid * w) & (W_ - 1);
        c += row[w] * ct[p];
        s += row[w] * st[p];
    }
    cbuf[tid] = c;
    sbuf[tid] = s;
    __syncthreads();

    const float inv = 1.0f / (float)W_;
    for (int i = 0; i < W_; ++i) {
        const int p = (tid * i) & (W_ - 1);
        const float v = (cbuf[i] * ct[p] + sbuf[i] * st[p]) * inv;
        X0pad[(((size_t)b * C0 + i) * PH + (h + 1)) * PW + (tid + 1)] = (__bf16)v;
    }
}

// ---------------------------------------------------------------------------
// mask_embedding (f32) -> X0pad channels 128..133 (bf16, halo'd)
// ---------------------------------------------------------------------------
__global__ __launch_bounds__(256)
void mask_embed_kernel(const float* __restrict__ me, __bf16* __restrict__ X0pad) {
    constexpr int PH = H_ + 2, PW = W_ + 2;
    const int idx = blockIdx.x * 256 + threadIdx.x;   // over B*EMB*H*W
    if (idx >= B_ * EMB * H_ * W_) return;
    const int w = idx % W_;
    const int h = (idx / W_) % H_;
    const int e = (idx / (H_ * W_)) % EMB;
    const int b = idx / (H_ * W_ * EMB);
    X0pad[(((size_t)b * C0 + (W_ + e)) * PH + (h + 1)) * PW + (w + 1)] = (__bf16)me[idx];
}

// ---------------------------------------------------------------------------
// Conv 4x4 / stride 2 / pad 1 as implicit GEMM, v_wmma_f32_16x16x32_bf16.
// Input is halo'd bf16 (B,Cin,IH+2,IW+2) -> completely branchless A gather:
// per lane 8 aligned b32 loads (4 runs of 4 contiguous bf16).  B fragment is
// one aligned 32B vector load from the pre-packed bf16 weight buffer.
// No LDS, no barriers, no conversions in the K loop.
// Block = 128 thr = 4 waves; wave w owns output tile (m0..m0+15) x 16 chans.
// ---------------------------------------------------------------------------
template <int Cin, int Cout, int IH, int IW, int OH, int OW, bool FUSE_LRELU_PAD>
__global__ __launch_bounds__(128)
void conv4x4s2_wmma(const __bf16* __restrict__ Xpad, const __bf16* __restrict__ Bp,
                    const float* __restrict__ bias,
                    float* __restrict__ Yf32, __bf16* __restrict__ Ypad) {
    constexpr int PH = IH + 2, PW = IW + 2;
    constexpr int KSTEPS = Cin / 2;          // 32-wide K chunks (2 cin planes)
    constexpr int rowU   = PW / 2;           // uints per padded input row
    constexpr int planeU = (PH * PW) / 2;    // uints per padded input plane

    const int lane = threadIdx.x & 31;
    const int wave = threadIdx.x >> 5;
    const int m0   = blockIdx.x * 16;
    const int n0   = blockIdx.y * 64 + wave * 16;

    // ---- per-lane A base address (WMMA A layout: lanes L and L+16 share row
    //      M=L&15; lane half selects cin plane 0 or 1, elements = kh*4+kw) ----
    {
    }
    const int ml = lane & 15;
    const int m  = m0 + ml;
    const int ow = m % OW;
    const int oh = (m / OW) % OH;
    const int bb = m / (OW * OH);
    const int cinH = lane >> 4;
    const unsigned int* ap = reinterpret_cast<const unsigned int*>(
        Xpad + (((size_t)bb * Cin + cinH) * PH + (size_t)(oh * 2)) * PW + (size_t)(ow * 2));
    const __bf16* bp = Bp + ((size_t)blockIdx.y * 4 + wave) * (size_t)KSTEPS * 512
                          + (size_t)lane * 16;

    v8f acc = {};

    #pragma unroll 2
    for (int ks = 0; ks < KSTEPS; ++ks) {
        v8u au;
        au[0] = ap[0];                 au[1] = ap[1];                  // cin cinH, row 0
        au[2] = ap[rowU];              au[3] = ap[rowU + 1];           // cin cinH, row 1
        au[4] = ap[2 * rowU];          au[5] = ap[2 * rowU + 1];       // cin cinH, row 2
        au[6] = ap[3 * rowU];          au[7] = ap[3 * rowU + 1];       // cin cinH, row 3
        const v16bf afrag = __builtin_bit_cast(v16bf, au);
        const v16bf bfrag = *reinterpret_cast<const v16bf*>(bp);
        acc = __builtin_amdgcn_wmma_f32_16x16x32_bf16(
            /*neg_a=*/false, afrag, /*neg_b=*/false, bfrag,
            /*c_mod=*/(short)0, acc, /*reuse_a=*/false, /*reuse_b=*/false);
        ap += planeU * 2;   // advance 2 cin planes
        bp += 512;          // next packed 32x16 B fragment
    }

    // ---- epilogue: D layout (VGPR r: lanes<16 -> M=r, lanes>=16 -> M=8+r) ----
    const int n  = n0 + (lane & 15);
    const float bn = bias[n];
    #pragma unroll
    for (int r = 0; r < 8; ++r) {
        const int mo  = m0 + r + ((lane >> 4) << 3);
        const int owo = mo % OW;
        const int oho = (mo / OW) % OH;
        const int bo  = mo / (OW * OH);
        float v = acc[r] + bn;
        if (FUSE_LRELU_PAD) {
            v = LRELU(v);
            Ypad[(((size_t)bo * Cout + n) * (OH + 2) + (oho + 1)) * (OW + 2) + (owo + 1)] =
                (__bf16)v;
        } else {
            Yf32[(((size_t)bo * Cout + n) * OH + oho) * OW + owo] = v;
        }
    }
}

// ---------------------------------------------------------------------------
// InstanceNorm2d (affine=false, eps 1e-5, biased var) + LeakyReLU.
// One block per (b,c) plane.  OUT_PAD_BF16 -> halo'd bf16 for the next conv,
// else plain f32 (last layer, feeds avg-pool).
// ---------------------------------------------------------------------------
template <int OH, int OW, bool OUT_PAD_BF16>
__global__ __launch_bounds__(256)
void inorm_lrelu_kernel(const float* __restrict__ Y,
                        __bf16* __restrict__ Apad, float* __restrict__ Af) {
    constexpr int SPATIAL = OH * OW;
    const int bc = blockIdx.x;
    const float* p = Y + (size_t)bc * SPATIAL;
    __shared__ float ssum[256], ssq[256];

    float s = 0.0f, q = 0.0f;
    for (int i = threadIdx.x; i < SPATIAL; i += 256) {
        const float v = p[i];
        s += v; q += v * v;
    }
    ssum[threadIdx.x] = s; ssq[threadIdx.x] = q;
    __syncthreads();
    #pragma unroll
    for (int st = 128; st > 0; st >>= 1) {
        if (threadIdx.x < st) {
            ssum[threadIdx.x] += ssum[threadIdx.x + st];
            ssq[threadIdx.x]  += ssq[threadIdx.x + st];
        }
        __syncthreads();
    }
    const float mean = ssum[0] / (float)SPATIAL;
    const float var  = ssq[0] / (float)SPATIAL - mean * mean;
    const float rstd = rsqrtf(var + 1e-5f);

    for (int i = threadIdx.x; i < SPATIAL; i += 256) {
        float v = (p[i] - mean) * rstd;
        v = LRELU(v);
        if (OUT_PAD_BF16) {
            const int ow = i % OW, oh = i / OW;
            Apad[((size_t)bc * (OH + 2) + (oh + 1)) * (OW + 2) + (ow + 1)] = (__bf16)v;
        } else {
            Af[(size_t)bc * SPATIAL + i] = v;
        }
    }
}

// ---------------------------------------------------------------------------
// 8x8 average pool -> pooled (B, 1024)
// ---------------------------------------------------------------------------
__global__ __launch_bounds__(256)
void avgpool_kernel(const float* __restrict__ A, float* __restrict__ pooled) {
    const int idx = blockIdx.x * 256 + threadIdx.x;   // over B*1024
    if (idx >= B_ * 1024) return;
    const float* p = A + (size_t)idx * 64;
    float s = 0.0f;
    #pragma unroll
    for (int i = 0; i < 64; ++i) s += p[i];
    pooled[idx] = s * (1.0f / 64.0f);
}

// ---------------------------------------------------------------------------
// Head: out[b,o] = pooled[b,:] . w4[o,:] + b4[o]
// ---------------------------------------------------------------------------
__global__ __launch_bounds__(256)
void head_kernel(const float* __restrict__ pooled, const float* __restrict__ w4,
                 const float* __restrict__ b4, float* __restrict__ out) {
    const int idx = blockIdx.x * 256 + threadIdx.x;   // over B*128
    if (idx >= B_ * 128) return;
    const int b = idx >> 7;
    const int o = idx & 127;
    const float* pv = pooled + (size_t)b * 1024;
    const float* wv = w4 + (size_t)o * 1024;
    float acc = 0.0f;
    #pragma unroll 8
    for (int c = 0; c < 1024; ++c) acc = fmaf(pv[c], wv[c], acc);
    out[idx] = acc + b4[o];
}

// ---------------------------------------------------------------------------
// Host launcher
// ---------------------------------------------------------------------------
extern "C" void kernel_launch(void* const* d_in, const int* in_sizes, int n_in,
                              void* d_out, int out_size, void* d_ws, size_t ws_size,
                              hipStream_t stream) {
    const float* image = (const float*)d_in[0];
    const float* me    = (const float*)d_in[1];
    const float* w0 = (const float*)d_in[2];  const float* b0 = (const float*)d_in[3];
    const float* w1 = (const float*)d_in[4];  const float* b1 = (const float*)d_in[5];
    const float* w2 = (const float*)d_in[6];  const float* b2 = (const float*)d_in[7];
    const float* w3 = (const float*)d_in[8];  const float* b3 = (const float*)d_in[9];
    const float* w4 = (const float*)d_in[10]; const float* b4 = (const float*)d_in[11];
    float* out = (float*)d_out;

    char* ws = (char*)d_ws;
    size_t off = 0;
    auto alloc = [&](size_t bytes) -> char* {
        char* p = ws + off;
        off += (bytes + 255) & ~(size_t)255;
        return p;
    };

    // Halo'd bf16 activation buffers (allocated first & contiguous -> one zero pass)
    __bf16* X0pad   = (__bf16*)alloc((size_t)B_ * C0   * 130 * 130 * 2);  // 36.2 MB
    __bf16* act0pad = (__bf16*)alloc((size_t)B_ * 256  *  66 *  66 * 2);  // 17.8 MB
    __bf16* act1pad = (__bf16*)alloc((size_t)B_ * 512  *  34 *  34 * 2);  //  9.5 MB
    __bf16* act2pad = (__bf16*)alloc((size_t)B_ * 1024 *  18 *  18 * 2);  //  5.3 MB
    const size_t zero_bytes = off;                                        // halo region

    // One f32 scratch reused by Y1/Y2/Y3 (disjoint lifetimes)
    float* Yscratch = (float*)alloc((size_t)B_ * 512 * 32 * 32 * 4);      // 16.8 MB
    float* act3     = (float*)alloc((size_t)B_ * 1024 * 8 * 8 * 4);       //  2.1 MB
    float* pooled   = (float*)alloc((size_t)B_ * 1024 * 4);

    // Packed bf16 weights (WMMA B-fragment layout)
    __bf16* Bp0 = (__bf16*)alloc((size_t)256  * 134  * 16 * 2);           //  1.1 MB
    __bf16* Bp1 = (__bf16*)alloc((size_t)512  * 256  * 16 * 2);           //  4.2 MB
    __bf16* Bp2 = (__bf16*)alloc((size_t)1024 * 512  * 16 * 2);           // 16.8 MB
    __bf16* Bp3 = (__bf16*)alloc((size_t)1024 * 1024 * 16 * 2);           // 33.6 MB

    // 0) zero the halo'd activation region (borders must be 0 every call)
    zero_kernel<<<4096, 256, 0, stream>>>((uint4*)ws, zero_bytes / 16);

    // 0b) pack weights to bf16 fragment layout
    pack_weights_kernel<<<(256  * 134  + 255) / 256, 256, 0, stream>>>(w0, Bp0, 134,  256  * 134);
    pack_weights_kernel<<<(512  * 256  + 255) / 256, 256, 0, stream>>>(w1, Bp1, 256,  512  * 256);
    pack_weights_kernel<<<(1024 * 512  + 255) / 256, 256, 0, stream>>>(w2, Bp2, 512,  1024 * 512);
    pack_weights_kernel<<<(1024 * 1024 + 255) / 256, 256, 0, stream>>>(w3, Bp3, 1024, 1024 * 1024);

    // 1) spectral map (channels 0..127) + mask embedding (128..133)
    spectral_map_kernel<<<B_ * H_, 128, 0, stream>>>(image, X0pad);
    mask_embed_kernel<<<(B_ * EMB * H_ * W_ + 255) / 256, 256, 0, stream>>>(me, X0pad);

    // 2) conv0: 134->256, 128x128 -> 64x64, fused lrelu, halo'd bf16 out
    conv4x4s2_wmma<134, 256, 128, 128, 64, 64, true>
        <<<dim3((B_ * 64 * 64) / 16, 256 / 64), 128, 0, stream>>>(
            X0pad, Bp0, b0, nullptr, act0pad);

    // 3) conv1: 256->512, 64x64 -> 32x32; inorm+lrelu -> halo'd bf16
    conv4x4s2_wmma<256, 512, 64, 64, 32, 32, false>
        <<<dim3((B_ * 32 * 32) / 16, 512 / 64), 128, 0, stream>>>(
            act0pad, Bp1, b1, Yscratch, nullptr);
    inorm_lrelu_kernel<32, 32, true><<<B_ * 512, 256, 0, stream>>>(Yscratch, act1pad, nullptr);

    // 4) conv2: 512->1024, 32x32 -> 16x16
    conv4x4s2_wmma<512, 1024, 32, 32, 16, 16, false>
        <<<dim3((B_ * 16 * 16) / 16, 1024 / 64), 128, 0, stream>>>(
            act1pad, Bp2, b2, Yscratch, nullptr);
    inorm_lrelu_kernel<16, 16, true><<<B_ * 1024, 256, 0, stream>>>(Yscratch, act2pad, nullptr);

    // 5) conv3: 1024->1024, 16x16 -> 8x8
    conv4x4s2_wmma<1024, 1024, 16, 16, 8, 8, false>
        <<<dim3((B_ * 8 * 8) / 16, 1024 / 64), 128, 0, stream>>>(
            act2pad, Bp3, b3, Yscratch, nullptr);
    inorm_lrelu_kernel<8, 8, false><<<B_ * 1024, 256, 0, stream>>>(Yscratch, nullptr, act3);

    // 6) avgpool + head
    avgpool_kernel<<<(B_ * 1024 + 255) / 256, 256, 0, stream>>>(act3, pooled);
    head_kernel<<<(B_ * 128 + 255) / 256, 256, 0, stream>>>(pooled, w4, b4, out);
}
